// GroupedQueryAttention_43267500540434
// MI455X (gfx1250) — compile-verified, matrix operand order checked by reading, not run on hardware
//
#include <hip/hip_runtime.h>
#include <hip/hip_bf16.h>

// ---------------------------------------------------------------------------
// GQA forward for MI455X (gfx1250, wave32, WMMA, async-LDS copies).
//   B=2, S=2048, D=2048, NH=32, NKV=8, HD=64, GS=4
// Pipeline:
//   pack x -> bf16, transpose-pack weights -> bf16 [N][K]
//   Q/K/V projection GEMMs (bf16 WMMA, f32 accum)
//   RoPE -> bf16 Q (pre-scaled 1/sqrt(HD)) and bf16 K
//   V transposed-packed to bf16 [B*KVD][S] (keys contiguous per hd row)
//   causal flash-attention (bf16 WMMA) -> bf16 ctx
//   output GEMM -> f32 out
// ---------------------------------------------------------------------------

typedef __attribute__((ext_vector_type(16))) __bf16 v16bf;
typedef __attribute__((ext_vector_type(8)))  __bf16 v8bf;
typedef __attribute__((ext_vector_type(4)))  __bf16 v4bf;
typedef __attribute__((ext_vector_type(8)))  float  v8f;
typedef int v4i_gcc __attribute__((vector_size(4 * sizeof(int))));

static constexpr int Bv    = 2;
static constexpr int Sv    = 2048;
static constexpr int Dv    = 2048;
static constexpr int NHv   = 32;
static constexpr int NKVv  = 8;
static constexpr int HDv   = 64;
static constexpr int GSv   = 4;
static constexpr int KVD   = NKVv * HDv;   // 512
static constexpr int Mrows = Bv * Sv;      // 4096

// --- CDNA5 async global->LDS copy (guarded; falls back to b128 ld/st) ------
#if defined(__has_builtin)
#  if __has_builtin(__builtin_amdgcn_global_load_async_to_lds_b128)
#    define HAVE_ASYNC_LDS 1
#  endif
#endif
#ifndef HAVE_ASYNC_LDS
#  define HAVE_ASYNC_LDS 0
#endif

#if HAVE_ASYNC_LDS
typedef __attribute__((address_space(1))) v4i_gcc glob_v4i;
typedef __attribute__((address_space(3))) v4i_gcc lds_v4i;
#endif

__device__ __forceinline__ void cp16_g2l(const __bf16* g, __bf16* l) {
#if HAVE_ASYNC_LDS
  __builtin_amdgcn_global_load_async_to_lds_b128(
      (glob_v4i*)g, (lds_v4i*)l, /*offset=*/0, /*cpol=*/0);
#else
  *(v8bf*)l = *(const v8bf*)g;
#endif
}

__device__ __forceinline__ void async_copies_done() {
#if HAVE_ASYNC_LDS
#  if defined(__has_builtin) && __has_builtin(__builtin_amdgcn_s_wait_asynccnt)
  __builtin_amdgcn_s_wait_asynccnt(0);
#  else
  asm volatile("s_wait_asynccnt 0x0" ::: "memory");
#  endif
#endif
}

__device__ __forceinline__ v8f wmma_bf16(v16bf a, v16bf b, v8f c) {
  return __builtin_amdgcn_wmma_f32_16x16x32_bf16(
      false, a, false, b, (short)0, c, false, false);
}

// 16-element bf16 fragment from two 16-byte chunks (LDS or global).
__device__ __forceinline__ v16bf ld16(const __bf16* p0, const __bf16* p1) {
  v8bf lo = *(const v8bf*)p0;
  v8bf hi = *(const v8bf*)p1;
  v16bf r;
#pragma unroll
  for (int i = 0; i < 8; ++i) { r[i] = lo[i]; r[8 + i] = hi[i]; }
  return r;
}

// ---------------------------------------------------------------------------
// Elementwise f32 -> bf16 pack (n must be a multiple of 4).
// ---------------------------------------------------------------------------
__global__ __launch_bounds__(256)
void pack_bf16(const float* __restrict__ in, __bf16* __restrict__ out, int n4) {
  int i = blockIdx.x * blockDim.x + threadIdx.x;
  if (i >= n4) return;
  float4 v = ((const float4*)in)[i];
  v4bf o;
  o[0] = (__bf16)v.x; o[1] = (__bf16)v.y; o[2] = (__bf16)v.z; o[3] = (__bf16)v.w;
  ((v4bf*)out)[i] = o;
}

// ---------------------------------------------------------------------------
// Tiled transpose + pack: f32 in[rows][cols] -> bf16 out[cols][rows].
// rows, cols multiples of 32. 256 threads, 32x32 tile.
// ---------------------------------------------------------------------------
__global__ __launch_bounds__(256)
void transpose_pack(const float* __restrict__ in, __bf16* __restrict__ out,
                    int rows, int cols) {
  __shared__ float t[32][33];
  const int tx = threadIdx.x & 31;
  const int ty = threadIdx.x >> 5;        // 0..7
  const int c0 = blockIdx.x * 32;
  const int r0 = blockIdx.y * 32;
#pragma unroll
  for (int i = 0; i < 4; ++i)
    t[ty + i * 8][tx] = in[(size_t)(r0 + ty + i * 8) * cols + c0 + tx];
  __syncthreads();
#pragma unroll
  for (int i = 0; i < 4; ++i)
    out[(size_t)(c0 + ty + i * 8) * rows + r0 + tx] = (__bf16)t[tx][ty + i * 8];
}

// ---------------------------------------------------------------------------
// GEMM: C[M,N] (f32) = A[M,K] (bf16, row-major) @ Bt[N,K]^T (bf16, row-major
// along K). Block 256 threads (8 waves), tile 128x64, K-step 64.
// Staging via async global->LDS b128 copies when available.
// ---------------------------------------------------------------------------
__global__ __launch_bounds__(256)
void gemm_bf16_wmma(const __bf16* __restrict__ A, const __bf16* __restrict__ Bt,
                    float* __restrict__ C, int M, int N, int K) {
  constexpr int BM = 128, BN = 64, BK = 64;
  constexpr int ASTR = BK + 8;                  // padded to break bank strides
  constexpr int BSTR = BK + 8;
  __shared__ alignas(16) __bf16 As[BM * ASTR];  // 18.0 KB
  __shared__ alignas(16) __bf16 Bs[BN * BSTR];  //  9.0 KB

  const int tid  = threadIdx.x;
  const int w    = tid >> 5;
  const int lane = tid & 31;
  const int wr   = w & 3;
  const int wc   = w >> 2;
  const int bm   = blockIdx.y * BM;
  const int bn   = blockIdx.x * BN;

  const int mcol = lane & 15;
  const int half = lane >> 4;
  const int kk0  = half * 8;

  v8f acc[2][2] = {};

  for (int k0 = 0; k0 < K; k0 += BK) {
#if HAVE_ASYNC_LDS
#pragma unroll
    for (int i = 0; i < 4; ++i) {            // A: 1024 16B chunks / 256 thr
      int f = tid + i * 256, row = f >> 3, cc = (f & 7) * 8;
      cp16_g2l(A + (size_t)(bm + row) * K + k0 + cc, &As[row * ASTR + cc]);
    }
#pragma unroll
    for (int i = 0; i < 2; ++i) {            // B: 512 16B chunks / 256 thr
      int f = tid + i * 256, row = f >> 3, cc = (f & 7) * 8;
      cp16_g2l(Bt + (size_t)(bn + row) * K + k0 + cc, &Bs[row * BSTR + cc]);
    }
    async_copies_done();
#else
    v8bf ra[4], rb[2];
#pragma unroll
    for (int i = 0; i < 4; ++i) {
      int f = tid + i * 256, row = f >> 3, cc = (f & 7) * 8;
      ra[i] = *(const v8bf*)(A + (size_t)(bm + row) * K + k0 + cc);
    }
#pragma unroll
    for (int i = 0; i < 2; ++i) {
      int f = tid + i * 256, row = f >> 3, cc = (f & 7) * 8;
      rb[i] = *(const v8bf*)(Bt + (size_t)(bn + row) * K + k0 + cc);
    }
#pragma unroll
    for (int i = 0; i < 4; ++i) {
      int f = tid + i * 256, row = f >> 3, cc = (f & 7) * 8;
      *(v8bf*)&As[row * ASTR + cc] = ra[i];
    }
#pragma unroll
    for (int i = 0; i < 2; ++i) {
      int f = tid + i * 256, row = f >> 3, cc = (f & 7) * 8;
      *(v8bf*)&Bs[row * BSTR + cc] = rb[i];
    }
#endif
    if (k0 + BK < K) {                       // emits global_prefetch_b8
      __builtin_prefetch(A + (size_t)(bm + (tid >> 1)) * K + k0 + BK, 0, 1);
      __builtin_prefetch(Bt + (size_t)(bn + (tid >> 2)) * K + k0 + BK, 0, 1);
    }
    __syncthreads();

#pragma unroll
    for (int kk = 0; kk < BK; kk += 32) {
      v16bf a0 = ld16(&As[(wr * 32 + mcol) * ASTR + kk + kk0],
                      &As[(wr * 32 + mcol) * ASTR + kk + kk0 + 16]);
      v16bf a1 = ld16(&As[(wr * 32 + 16 + mcol) * ASTR + kk + kk0],
                      &As[(wr * 32 + 16 + mcol) * ASTR + kk + kk0 + 16]);
      v16bf b0 = ld16(&Bs[(wc * 32 + mcol) * BSTR + kk + kk0],
                      &Bs[(wc * 32 + mcol) * BSTR + kk + kk0 + 16]);
      v16bf b1 = ld16(&Bs[(wc * 32 + 16 + mcol) * BSTR + kk + kk0],
                      &Bs[(wc * 32 + 16 + mcol) * BSTR + kk + kk0 + 16]);
      acc[0][0] = wmma_bf16(a0, b0, acc[0][0]);
      acc[0][1] = wmma_bf16(a0, b1, acc[0][1]);
      acc[1][0] = wmma_bf16(a1, b0, acc[1][0]);
      acc[1][1] = wmma_bf16(a1, b1, acc[1][1]);
    }
    __syncthreads();
  }

#pragma unroll
  for (int i = 0; i < 2; ++i)
#pragma unroll
    for (int j = 0; j < 2; ++j)
#pragma unroll
      for (int r = 0; r < 8; ++r) {
        int row = bm + wr * 32 + i * 16 + r + half * 8;
        int col = bn + wc * 32 + j * 16 + mcol;
        C[(size_t)row * N + col] = acc[i][j][r];
      }
}

// ---------------------------------------------------------------------------
// RoPE + pack to bf16 (optionally pre-scaled). in f32 [B*S][H*64].
// ---------------------------------------------------------------------------
__global__ __launch_bounds__(256)
void rope_pack(const float* __restrict__ in, __bf16* __restrict__ out,
               int H, float scale, int total_pairs) {
  int p = blockIdx.x * blockDim.x + threadIdx.x;
  if (p >= total_pairs) return;
  int per_row = H * 32;
  int row  = p / per_row;
  int rem  = p - row * per_row;
  int head = rem >> 5;
  int j    = rem & 31;
  int pos  = row & (Sv - 1);
  float inv_freq = __powf(10000.0f, -(float)j * (1.0f / 32.0f));
  float sn, cs;
  __sincosf((float)pos * inv_freq, &sn, &cs);
  size_t base = (size_t)row * (H * 64) + head * 64;
  float x1 = in[base + j];
  float x2 = in[base + 32 + j];
  out[base + j]      = (__bf16)((x1 * cs - x2 * sn) * scale);
  out[base + 32 + j] = (__bf16)((x2 * cs + x1 * sn) * scale);
}

// ---------------------------------------------------------------------------
// Causal GQA flash-attention, all-bf16 operands.
//   Q  : bf16 [B*S][D]        (RoPE'd, pre-scaled by 1/sqrt(HD))
//   K  : bf16 [B*S][KVD]      (RoPE'd)
//   Vt : bf16 [B*KVD][S]      (transposed: keys contiguous per hd row)
//   ctx: bf16 [B*S][D]
// grid (S/128, B*NH), 256 threads; each wave owns 16 queries.
// ---------------------------------------------------------------------------
__global__ __launch_bounds__(256)
void gqa_flash_attn(const __bf16* __restrict__ Q, const __bf16* __restrict__ Kb,
                    const __bf16* __restrict__ Vt, __bf16* __restrict__ ctx) {
  constexpr int PSTR = 40;                      // [16 q][32 k] + pad
  __shared__ alignas(16) __bf16 Pb[8][16 * PSTR];

  const int tid  = threadIdx.x;
  const int w    = tid >> 5;
  const int lane = tid & 31;
  const int m    = lane & 15;
  const int half = lane >> 4;
  const int k0   = half * 8;

  const int bh  = blockIdx.y;
  const int b   = bh / NHv;
  const int h   = bh - b * NHv;
  const int kvh = h / GSv;
  const int qbase = blockIdx.x * 128 + w * 16;  // wave-uniform

  const __bf16* qrow = Q + (size_t)(b * Sv + qbase + m) * Dv + h * HDv;
  const v16bf qf0 = ld16(qrow + k0,      qrow + k0 + 16);
  const v16bf qf1 = ld16(qrow + 32 + k0, qrow + 48 + k0);

  v8f   Oacc[4] = {};
  float mrun[8], lrun[8];
#pragma unroll
  for (int r = 0; r < 8; ++r) { mrun[r] = -1e30f; lrun[r] = 0.0f; }

  const int nchunks = (qbase + 47) >> 5;        // causal upper bound
  for (int kc = 0; kc < nchunks; ++kc) {
    const int kstart = kc * 32;

    // ---- QK^T: two 16x16 score fragments over HD=64
    v8f s0 = {}, s1 = {};
    {
      const __bf16* kr0 = Kb + (size_t)(b * Sv + kstart + m) * KVD + kvh * HDv;
      s0 = wmma_bf16(qf0, ld16(kr0 + k0, kr0 + k0 + 16), s0);
      s0 = wmma_bf16(qf1, ld16(kr0 + 32 + k0, kr0 + 48 + k0), s0);
      const __bf16* kr1 = Kb + (size_t)(b * Sv + kstart + 16 + m) * KVD + kvh * HDv;
      s1 = wmma_bf16(qf0, ld16(kr1 + k0, kr1 + k0 + 16), s1);
      s1 = wmma_bf16(qf1, ld16(kr1 + 32 + k0, kr1 + 48 + k0), s1);
    }

    // ---- causal mask + chunk row-max (16-lane butterfly)
    float rmax[8];
#pragma unroll
    for (int r = 0; r < 8; ++r) {
      int qidx = qbase + r + half * 8;
      float a  = ((kstart + m)      > qidx) ? -1e30f : s0[r];
      float bb = ((kstart + 16 + m) > qidx) ? -1e30f : s1[r];
      s0[r] = a; s1[r] = bb;
      rmax[r] = fmaxf(a, bb);
    }
#pragma unroll
    for (int mask = 1; mask <= 8; mask <<= 1)
#pragma unroll
      for (int r = 0; r < 8; ++r)
        rmax[r] = fmaxf(rmax[r], __shfl_xor(rmax[r], mask, 32));

    // ---- online softmax; stash P (bf16) into per-wave LDS
    float alpha[8], rsum[8];
#pragma unroll
    for (int r = 0; r < 8; ++r) {
      float mn = fmaxf(mrun[r], rmax[r]);
      alpha[r] = __expf(mrun[r] - mn);
      mrun[r]  = mn;
      float p0 = __expf(s0[r] - mn);
      float p1 = __expf(s1[r] - mn);
      rsum[r] = p0 + p1;
      Pb[w][(r + half * 8) * PSTR + m]      = (__bf16)p0;
      Pb[w][(r + half * 8) * PSTR + 16 + m] = (__bf16)p1;
    }
#pragma unroll
    for (int mask = 1; mask <= 8; mask <<= 1)
#pragma unroll
      for (int r = 0; r < 8; ++r)
        rsum[r] += __shfl_xor(rsum[r], mask, 32);
#pragma unroll
    for (int r = 0; r < 8; ++r) lrun[r] = lrun[r] * alpha[r] + rsum[r];
#pragma unroll
    for (int c = 0; c < 4; ++c)
#pragma unroll
      for (int r = 0; r < 8; ++r) Oacc[c][r] *= alpha[r];

    // ---- P(16x32) @ V(32x64): P from LDS, V straight from global (bf16,
    //      key-contiguous thanks to the global transpose)
    v16bf pa = ld16(&Pb[w][m * PSTR + k0], &Pb[w][m * PSTR + k0 + 16]);
#pragma unroll
    for (int c = 0; c < 4; ++c) {
      const __bf16* vr =
          Vt + (size_t)(b * KVD + kvh * HDv + c * 16 + m) * Sv + kstart;
      Oacc[c] = wmma_bf16(pa, ld16(vr + k0, vr + k0 + 16), Oacc[c]);
    }
  }

  // ---- normalize, write bf16 ctx [B*S][D]
  float linv[8];
#pragma unroll
  for (int r = 0; r < 8; ++r) linv[r] = 1.0f / lrun[r];
#pragma unroll
  for (int c = 0; c < 4; ++c)
#pragma unroll
    for (int r = 0; r < 8; ++r) {
      int qidx = qbase + r + half * 8;
      ctx[(size_t)(b * Sv + qidx) * Dv + h * HDv + c * 16 + m] =
          (__bf16)(Oacc[c][r] * linv[r]);
    }
}

// ---------------------------------------------------------------------------
// Launch. Workspace layout (bytes, 256-aligned):
//   xbf 16M | Wqt 8M | Wkt 2M | Wvt 2M | Wot 8M | Qf 32M | Kf 8M | Vf 8M
//   Qbf 16M | Kbf 4M | Vtbf 4M | ctxbf 16M   (~124 MB)
// ---------------------------------------------------------------------------
extern "C" void kernel_launch(void* const* d_in, const int* in_sizes, int n_in,
                              void* d_out, int out_size, void* d_ws, size_t ws_size,
                              hipStream_t stream) {
  (void)in_sizes; (void)n_in; (void)out_size; (void)ws_size;
  const float* x  = (const float*)d_in[0];
  const float* Wq = (const float*)d_in[1];
  const float* Wk = (const float*)d_in[2];
  const float* Wv = (const float*)d_in[3];
  const float* Wo = (const float*)d_in[4];
  float* out = (float*)d_out;

  char*  ws  = (char*)d_ws;
  size_t off = 0;
  auto carve = [&](size_t bytes) {
    void* p = ws + off;
    off += (bytes + 255) & ~(size_t)255;
    return p;
  };
  __bf16* xbf   = (__bf16*)carve((size_t)Mrows * Dv * 2);
  __bf16* Wqt   = (__bf16*)carve((size_t)Dv * Dv * 2);
  __bf16* Wkt   = (__bf16*)carve((size_t)KVD * Dv * 2);
  __bf16* Wvt   = (__bf16*)carve((size_t)KVD * Dv * 2);
  __bf16* Wot   = (__bf16*)carve((size_t)Dv * Dv * 2);
  float*  Qf    = (float*)carve((size_t)Mrows * Dv * 4);
  float*  Kf    = (float*)carve((size_t)Mrows * KVD * 4);
  float*  Vf    = (float*)carve((size_t)Mrows * KVD * 4);
  __bf16* Qbf   = (__bf16*)carve((size_t)Mrows * Dv * 2);
  __bf16* Kbf   = (__bf16*)carve((size_t)Mrows * KVD * 2);
  __bf16* Vtbf  = (__bf16*)carve((size_t)Bv * KVD * Sv * 2);
  __bf16* ctxbf = (__bf16*)carve((size_t)Mrows * Dv * 2);

  dim3 blk(256);

  // --- pack inputs to bf16 (weights transposed to [N][K])
  {
    int n4 = Mrows * Dv / 4;
    pack_bf16<<<(n4 + 255) / 256, blk, 0, stream>>>(x, xbf, n4);
  }
  transpose_pack<<<dim3(Dv / 32,  Dv / 32), blk, 0, stream>>>(Wq, Wqt, Dv, Dv);
  transpose_pack<<<dim3(KVD / 32, Dv / 32), blk, 0, stream>>>(Wk, Wkt, Dv, KVD);
  transpose_pack<<<dim3(KVD / 32, Dv / 32), blk, 0, stream>>>(Wv, Wvt, Dv, KVD);
  transpose_pack<<<dim3(Dv / 32,  Dv / 32), blk, 0, stream>>>(Wo, Wot, Dv, Dv);

  // --- projections (f32 accum out)
  gemm_bf16_wmma<<<dim3(Dv / 64,  Mrows / 128), blk, 0, stream>>>(xbf, Wqt, Qf, Mrows, Dv,  Dv);
  gemm_bf16_wmma<<<dim3(KVD / 64, Mrows / 128), blk, 0, stream>>>(xbf, Wkt, Kf, Mrows, KVD, Dv);
  gemm_bf16_wmma<<<dim3(KVD / 64, Mrows / 128), blk, 0, stream>>>(xbf, Wvt, Vf, Mrows, KVD, Dv);

  // --- RoPE (+1/sqrt(HD) into Q) and V transpose, all to bf16
  const int qPairs = Mrows * NHv  * 32;
  const int kPairs = Mrows * NKVv * 32;
  rope_pack<<<(qPairs + 255) / 256, blk, 0, stream>>>(Qf, Qbf, NHv,  0.125f, qPairs);
  rope_pack<<<(kPairs + 255) / 256, blk, 0, stream>>>(Kf, Kbf, NKVv, 1.0f,   kPairs);
  for (int b = 0; b < Bv; ++b)
    transpose_pack<<<dim3(KVD / 32, Sv / 32), blk, 0, stream>>>(
        Vf + (size_t)b * Sv * KVD, Vtbf + (size_t)b * KVD * Sv, Sv, KVD);

  // --- attention -> bf16 ctx
  gqa_flash_attn<<<dim3(Sv / 128, Bv * NHv), blk, 0, stream>>>(Qbf, Kbf, Vtbf, ctxbf);

  // --- output projection -> f32 out
  gemm_bf16_wmma<<<dim3(Dv / 64, Mrows / 128), blk, 0, stream>>>(ctxbf, Wot, out, Mrows, Dv, Dv);
}
